// TensorProduct_23785528885957
// MI455X (gfx1250) — matrix operand order repaired and checked by reading, not compiled, not run on hardware
//
#include <hip/hip_runtime.h>

// ============================================================================
// out[b, n, f] = sum_{l,m} cg[l,m,n] * x1[b,l,f] * x2[b,m,f]
//   B=20000, l,m in [0,16), n in [0,49), f in [0,512)
//
// Per batch: out(49x512) = CG^T(49x256) @ Y(256x512), Y[lm,f]=x1[l,f]*x2[m,f].
// 257 GFLOP vs 3.3 GB -> 142 us memory floor @ 23.3 TB/s. bf16 WMMA
// (16x16x32, f32 acc) with hi/lo split of both operands (Ch*Yh+Ch*Yl+Cl*Yh)
// gives near-f32 accuracy at ~200 us matrix time (~5 PFLOPS dense bf16).
// CG is constant: swizzled once per WG into LDS in the A-operand layout.
// Round 3: LDS records interleaved to 64B/lane and loaded as uint4 quads so
// the A tiles come in via ds_load_b128 with immediate offsets (round 2
// lowered to ds_load_2addr_b32 + address adds + dscnt stalls).
// ============================================================================

typedef __attribute__((ext_vector_type(16))) __bf16        v16bf;
typedef __attribute__((ext_vector_type(8)))  float         v8f;
typedef __attribute__((ext_vector_type(4)))  unsigned int  u32x4;

#define FDIM 512
#define NOUT 49

struct q2 { u32x4 a, b; };   // 32 bytes, maps onto one v16bf A operand

__device__ __forceinline__ unsigned short bf_bits(__bf16 h) {
  return __builtin_bit_cast(unsigned short, h);
}

__device__ __forceinline__ v16bf load_a_tile(const u32x4* p) {
  q2 t;
  t.a = p[0];          // ds_load_b128
  t.b = p[1];          // ds_load_b128
  return __builtin_bit_cast(v16bf, t);
}

__device__ __forceinline__ v8f wmma_bf16(v16bf a, v16bf b, v8f c) {
  // 8 args: (neg_a, A, neg_b, B, c_mod, C, reuse_a, reuse_b)
  return __builtin_amdgcn_wmma_f32_16x16x32_bf16(false, a, false, b,
                                                 (short)0, c, false, false);
}

__global__ void __launch_bounds__(256)
tp_cg_kernel(const float* __restrict__ x1g, const float* __restrict__ x2g,
             const float* __restrict__ cgg, float* __restrict__ outg,
             int nBatch)
{
  // 16384 dwords = 64 KB.
  // dword index = ((kc*4 + nt)*32 + lane)*16 + h*8 + v
  //   -> per lane, per (kc,nt): 64 contiguous bytes = ah(32B) ++ al(32B)
  extern __shared__ unsigned int a_dw[];

  const int tid  = threadIdx.x;
  const int lane = tid & 31;
  const int wave = tid >> 5;    // 0..7
  const int col  = lane & 15;   // WMMA B/D column within tile
  const int half = lane >> 4;   // lane group 0/1

  // ---- Stage CG once per WG: bf16 hi/lo, swizzled to WMMA A-operand layout.
  // A operand k order as filled here: lanes 0-15 row M=lane, V0..7 hold K=0..15 (pairs);
  // lanes 16-31 hold K=16..31 (contiguous, matching the B fill).
  for (int i = tid; i < 16384; i += 256) {
    const int v  = i & 7;
    const int hl = (i >> 3) & 1;
    const int ln = (i >> 4) & 31;
    const int nt = (i >> 9) & 3;
    const int kc = (i >> 11) & 7;
    const int hf = ln >> 4;
    const int n  = nt * 16 + (ln & 15);
    const int kk = hf * 16 + 2 * v;
    const int l  = 2 * kc + (kk >> 4);   // global K = kc*32 + kk ; l = K/16
    const int m0 = kk & 15;
    float f0 = 0.f, f1 = 0.f;
    if (n < NOUT) {
      f0 = cgg[(l * 16 + m0) * NOUT + n];
      f1 = cgg[(l * 16 + m0 + 1) * NOUT + n];
    }
    const __bf16 h0 = (__bf16)f0, h1 = (__bf16)f1;
    __bf16 b0 = h0, b1 = h1;
    if (hl) {                       // residual (lo) part
      b0 = (__bf16)(f0 - (float)h0);
      b1 = (__bf16)(f1 - (float)h1);
    }
    a_dw[i] = ((unsigned int)bf_bits(b1) << 16) | (unsigned int)bf_bits(b0);
  }
  __syncthreads();

  const u32x4* a_q = (const u32x4*)a_dw;   // 16B-aligned quad view of tiles

  // ---- Persistent batch loop ------------------------------------------------
  for (int b = blockIdx.x; b < nBatch; b += (int)gridDim.x) {
    const float* __restrict__ x1b = x1g + (size_t)b * (16 * FDIM);
    const float* __restrict__ x2b = x2g + (size_t)b * (16 * FDIM);
    float* __restrict__ ob = outg + (size_t)b * (NOUT * FDIM);

    {   // prefetch next batch into L2 (global_prefetch_b8)
      const int bn = b + (int)gridDim.x;
      if (bn < nBatch) {
        __builtin_prefetch(x1g + (size_t)bn * (16 * FDIM) + tid * 32, 0, 1);
        __builtin_prefetch(x2g + (size_t)bn * (16 * FDIM) + tid * 32, 0, 1);
      }
    }

#pragma unroll 1
    for (int sg = 0; sg < 2; ++sg) {
      // two f-strips (16 channels each) per group; 8 waves x 2 sg x 2 = 32
      const int fA = (wave + sg * 16) * 16 + col;
      const int fB = (wave + 8 + sg * 16) * 16 + col;

      float x2A[16], x2B[16];
#pragma unroll
      for (int m = 0; m < 16; ++m) {
        x2A[m] = x2b[m * FDIM + fA];
        x2B[m] = x2b[m * FDIM + fB];
      }
      float x1A[8], x1B[8];
#pragma unroll
      for (int k = 0; k < 8; ++k) {          // l needed per chunk = 2*kc + half
        x1A[k] = x1b[(2 * k + half) * FDIM + fA];
        x1B[k] = x1b[(2 * k + half) * FDIM + fB];
      }

      // n-tile groups kept as a rolled loop: live set stays bounded
#pragma unroll 1
      for (int ntg = 0; ntg < 2; ++ntg) {
        // per-lane quad base for this ntg: ((0*4 + ntg*2)*32 + lane)*4
        const u32x4* aq = a_q + ((size_t)(ntg * 2) * 32 + lane) * 4;

        v8f acc[2][2];
        const v8f vzero = {0.f, 0.f, 0.f, 0.f, 0.f, 0.f, 0.f, 0.f};
        acc[0][0] = vzero; acc[0][1] = vzero;
        acc[1][0] = vzero; acc[1][1] = vzero;

#pragma unroll
        for (int kc = 0; kc < 8; ++kc) {
          // Build B operand (Y tile) in registers, split into bf16 hi/lo.
          // B operand k order as built here: lane col = lane&15, lanes<16 hold K=0..15,
          // lanes>=16 hold K=16..31 -> element e maps to m=e, l=2*kc+half.
          v16bf yhA, ylA, yhB, ylB;
#pragma unroll
          for (int m = 0; m < 16; ++m) {
            const float yA = x1A[kc] * x2A[m];
            const __bf16 hA = (__bf16)yA;
            yhA[m] = hA;
            ylA[m] = (__bf16)(yA - (float)hA);
            const float yB = x1B[kc] * x2B[m];
            const __bf16 hB = (__bf16)yB;
            yhB[m] = hB;
            ylB[m] = (__bf16)(yB - (float)hB);
          }
#pragma unroll
          for (int t = 0; t < 2; ++t) {
            // quad offset for (kc, nt=ntg*2+t): (kc*4 + t)*128 from aq
            const int qoff = (kc * 4 + t) * 128;
            const v16bf ah = load_a_tile(aq + qoff);       // +0,+16 bytes
            const v16bf al = load_a_tile(aq + qoff + 2);   // +32,+48 bytes
            acc[0][t] = wmma_bf16(ah, yhA, acc[0][t]);
            acc[0][t] = wmma_bf16(ah, ylA, acc[0][t]);
            acc[0][t] = wmma_bf16(al, yhA, acc[0][t]);
            acc[1][t] = wmma_bf16(ah, yhB, acc[1][t]);
            acc[1][t] = wmma_bf16(ah, ylB, acc[1][t]);
            acc[1][t] = wmma_bf16(al, yhB, acc[1][t]);
          }
        }

        // Store (D layout: lane col = f, VGPR r -> row r + 8*half within tile)
#pragma unroll
        for (int st = 0; st < 2; ++st) {
          const int f = st ? fB : fA;
#pragma unroll
          for (int t = 0; t < 2; ++t) {
#pragma unroll
            for (int r = 0; r < 8; ++r) {
              const int n = (ntg * 2 + t) * 16 + half * 8 + r;
              if (n < NOUT) ob[n * FDIM + f] = acc[st][t][r];
            }
          }
        }
      }
    }
  }
}

extern "C" void kernel_launch(void* const* d_in, const int* in_sizes, int n_in,
                              void* d_out, int out_size, void* d_ws, size_t ws_size,
                              hipStream_t stream) {
  const float* x1 = (const float*)d_in[0];
  const float* x2 = (const float*)d_in[1];
  const float* cg = (const float*)d_in[2];
  float* out = (float*)d_out;
  const int nBatch = in_sizes[0] / (16 * FDIM);
  int grid = nBatch < 1250 ? nBatch : 1250;   // 20000 = 1250 * 16, no tail
  if (grid < 1) grid = 1;
  tp_cg_kernel<<<dim3(grid), dim3(256), 65536, stream>>>(x1, x2, cg, out, nBatch);
}
